// ScaleAttention_84731114816130
// MI455X (gfx1250) — compile-verified, matrix operand order checked by reading, not run on hardware
//
#include <hip/hip_runtime.h>

// CDNA5 / gfx1250 fused masked-softmax attention forward.
// out  = softmax(masked_fill(QK^T/temp, mask, 1e-9)) * mask @ V
// Returns (out, atten) -> d_out = [out | atten] concatenated.

typedef __attribute__((ext_vector_type(16))) __bf16 v16bf;
typedef __attribute__((ext_vector_type(8)))  float  v8f;

#define B_  2
#define H_  16
#define S_  2048
#define D_  128
#define INV_TEMP 0.08838834764831845f  // 1/sqrt(128)

__global__ __launch_bounds__(256, 2)
void attn_fwd_bf16wmma(const float* __restrict__ q,
                       const float* __restrict__ k,
                       const float* __restrict__ v,
                       const int*   __restrict__ mask,
                       float* __restrict__ out,
                       float* __restrict__ atten)
{
    const int bh   = blockIdx.x >> 7;     // 0..31  (b*H + h)
    const int mblk = blockIdx.x & 127;    // 0..127 (query block of 16 rows)
    const int b    = bh >> 4;
    const int row0 = mblk * 16;

    const int tid   = threadIdx.x;
    const int wave  = tid >> 5;           // 0..7
    const int lane  = tid & 31;
    const int arow  = lane & 15;          // A-fragment row this lane serves
    const int kbase = (lane >> 4) * 8;    // K sub-offset per 16-K group
    const int crow  = (lane >> 4) * 8;    // C/D fragment row base
    const int ccol  = lane & 15;          // C/D fragment column

    // ---- Load Q tile (16x128) as 4 bf16 A-fragments, pre-scaled by 1/temp ----
    const float* qrow = q + ((size_t)bh * S_ + row0 + arow) * D_;
    v16bf qa[4];
    #pragma unroll
    for (int kt = 0; kt < 4; ++kt) {
        #pragma unroll
        for (int e = 0; e < 8; ++e) {
            qa[kt][e]     = (__bf16)(qrow[kt * 32 + kbase + e]      * INV_TEMP);
            qa[kt][8 + e] = (__bf16)(qrow[kt * 32 + 16 + kbase + e] * INV_TEMP);
        }
    }

    float*     attnBlk = atten + ((size_t)bh * S_ + row0) * S_;  // [16][S_]
    const int* maskBlk = mask  + ((size_t)b  * S_ + row0) * S_;  // [16][S_] (H broadcast)

    // ---- Phase A: raw masked scores -> global atten region (scratch use) ----
    for (int t = 0; t < 16; ++t) {
        const int col0 = (t * 8 + wave) * 16;
        const float* krow = k + ((size_t)bh * S_ + col0 + arow) * D_;
        v8f acc = {0.f, 0.f, 0.f, 0.f, 0.f, 0.f, 0.f, 0.f};
        #pragma unroll
        for (int kt = 0; kt < 4; ++kt) {
            v16bf kb;
            #pragma unroll
            for (int e = 0; e < 8; ++e) {
                kb[e]     = (__bf16)krow[kt * 32 + kbase + e];
                kb[8 + e] = (__bf16)krow[kt * 32 + 16 + kbase + e];
            }
            acc = __builtin_amdgcn_wmma_f32_16x16x32_bf16(
                      false, qa[kt], false, kb, (short)0, acc, false, false);
        }
        #pragma unroll
        for (int i = 0; i < 8; ++i) {
            const int r = crow + i;
            const int m = maskBlk[(size_t)r * S_ + col0 + ccol];
            float s = acc[i];
            if (m) s = 1e-9f;   // masked_fill(mask.bool(), 1e-9)
            attnBlk[(size_t)r * S_ + col0 + ccol] = s;
        }
    }
    __threadfence_block();
    __syncthreads();

    // ---- Phase B: row softmax (16 threads per row), multiply by mask ----
    __shared__ float sPart[256];
    const int r  = tid >> 4;    // row 0..15
    const int c0 = tid & 15;    // column phase
    float*     prow16 = attnBlk + (size_t)r * S_;
    const int* mrow16 = maskBlk + (size_t)r * S_;

    float mx = -3.4e38f;
    for (int j = 0; j < S_ / 16; ++j)
        mx = fmaxf(mx, prow16[c0 + j * 16]);
    sPart[tid] = mx;
    __syncthreads();
    float rmax = sPart[r * 16];
    #pragma unroll
    for (int j = 1; j < 16; ++j) rmax = fmaxf(rmax, sPart[r * 16 + j]);
    __syncthreads();

    float sum = 0.f;
    for (int j = 0; j < S_ / 16; ++j) {
        const int c = c0 + j * 16;
        const float e = __expf(prow16[c] - rmax);
        prow16[c] = e;          // stash exp for normalize pass (same thread)
        sum += e;
    }
    sPart[tid] = sum;
    __syncthreads();
    float rsum = 0.f;
    #pragma unroll
    for (int j = 0; j < 16; ++j) rsum += sPart[r * 16 + j];
    const float inv = 1.0f / rsum;

    for (int j = 0; j < S_ / 16; ++j) {
        const int c = c0 + j * 16;
        float p = prow16[c] * inv;
        p = mrow16[c] ? p : 0.0f;   // atten *= mask (zero where mask==0)
        prow16[c] = p;              // final atten output AND P operand for PV
    }
    __threadfence_block();
    __syncthreads();

    // ---- Phase C: out tile (16 x 16) per wave = P @ V over K = 2048 ----
    v8f oacc = {0.f, 0.f, 0.f, 0.f, 0.f, 0.f, 0.f, 0.f};
    const float* vbase = v + (size_t)bh * S_ * D_ + wave * 16 + arow; // col = wave*16 + (lane&15)
    const float* parow = attnBlk + (size_t)arow * S_;
    for (int kt = 0; kt < 64; ++kt) {
        const int k0 = kt * 32;
        v16bf pa, vb;
        #pragma unroll
        for (int e = 0; e < 8; ++e) {
            pa[e]     = (__bf16)parow[k0 + kbase + e];
            pa[8 + e] = (__bf16)parow[k0 + 16 + kbase + e];
            vb[e]     = (__bf16)vbase[(size_t)(k0 + kbase + e) * D_];
            vb[8 + e] = (__bf16)vbase[(size_t)(k0 + 16 + kbase + e) * D_];
        }
        oacc = __builtin_amdgcn_wmma_f32_16x16x32_bf16(
                   false, pa, false, vb, (short)0, oacc, false, false);
    }
    #pragma unroll
    for (int i = 0; i < 8; ++i) {
        const int rr = crow + i;
        out[((size_t)bh * S_ + row0 + rr) * D_ + wave * 16 + ccol] = oacc[i];
    }
}

extern "C" void kernel_launch(void* const* d_in, const int* in_sizes, int n_in,
                              void* d_out, int out_size, void* d_ws, size_t ws_size,
                              hipStream_t stream) {
    (void)in_sizes; (void)n_in; (void)out_size; (void)d_ws; (void)ws_size;
    const float* q    = (const float*)d_in[0];
    const float* k    = (const float*)d_in[1];
    const float* v    = (const float*)d_in[2];
    const int*   mask = (const int*)d_in[3];

    float* out   = (float*)d_out;                              // [B,H,S,D]
    float* atten = out + (size_t)B_ * H_ * S_ * D_;            // [B,H,S,S]

    const int nblocks = B_ * H_ * (S_ / 16);                   // 4096
    attn_fwd_bf16wmma<<<nblocks, 256, 0, stream>>>(q, k, v, mask, out, atten);
}
